// EncoderBlock_81114752352790
// MI455X (gfx1250) — matrix operand order repaired and checked by reading, not compiled, not run on hardware
//
#include <hip/hip_runtime.h>
#include <stdint.h>

// ---------------------------------------------------------------------------
// Transformer encoder block for MI455X (gfx1250, wave32, WMMA).
// All matmuls through v_wmma_f32_16x16x32_bf16 (fp32 accum). Every GEMM B
// operand is kept [N,K] row-major so both LDS staging and per-lane fragment
// gathers are 16-byte vector ops. LDS tiles padded to 40-element pitch for
// bank-conflict-free b128 fragment loads.
// ---------------------------------------------------------------------------

#define Hh 8
#define Dd 512
#define DFFd 2048
#define Bb 8
#define Tt 1024
#define Mrows (Bb * Tt)   // 8192
#define LPAD 40           // LDS pitch (elements) for 128x32 tiles

typedef __attribute__((ext_vector_type(16))) __bf16 v16bf;
typedef __attribute__((ext_vector_type(8)))  float  v8f;

union AFrag { v16bf v; uint16_t u[16]; uint4 q[2]; };
union CFrag { v8f v; float f[8]; };
union Pack8 { uint16_t u[8]; uint4 q; };

__device__ __forceinline__ uint16_t f2bf(float x) {
  union { float f; uint32_t u; } v; v.f = x;
  uint32_t r = v.u + 0x7FFFu + ((v.u >> 16) & 1u);   // round-to-nearest-even
  return (uint16_t)(r >> 16);
}

// ---- WMMA fragment loaders (CDNA5 ISA 16-bit layouts) ----------
// A tile in LDS: [16+ rows][LPAD], row-major (m-major). Lane l: row = l&15,
// needs K = kb..kb+7 and kb+16..kb+23 (kb = (l&16)?8:0): two contiguous 16B.
__device__ __forceinline__ v16bf fragA(const uint16_t* t) {
  const int lane = threadIdx.x & 31;
  const uint16_t* p = t + (lane & 15) * LPAD + ((lane & 16) ? 8 : 0);
  AFrag a;
  a.q[0] = *(const uint4*)p;
  a.q[1] = *(const uint4*)(p + 16);
  return a.v;
}
// B tile in LDS: [16+ cols][LPAD], n-major. Lane l: col = l&15, needs
// K = kb..kb+7 and kb+16..kb+23 (kb = (l&16)?8:0): two contiguous 16B.
__device__ __forceinline__ v16bf fragB(const uint16_t* t) {
  const int lane = threadIdx.x & 31;
  const uint16_t* p = t + (lane & 15) * LPAD + ((lane & 16) ? 8 : 0);
  AFrag b;
  b.q[0] = *(const uint4*)p;
  b.q[1] = *(const uint4*)(p + 16);
  return b.v;
}

__device__ __forceinline__ float block_reduce_sum(float v, float* red) {
  const int tid = threadIdx.x;
  red[tid] = v; __syncthreads();
#pragma unroll
  for (int off = 128; off > 0; off >>= 1) {
    if (tid < off) red[tid] += red[tid + off];
    __syncthreads();
  }
  const float r = red[0];
  __syncthreads();
  return r;
}

// ---------------------------------------------------------------------------
// C[M,N] = scale * A[M,K] @ Bt[N,K]^T + bias.   A row-major, Bt [N,K]-major.
// 128x128x32 block tile, 8 waves, each 32x64 (2x4 WMMA tiles), double-
// buffered LDS. OUT_TRANS: store bf16 C transposed ([N,M], vectorized).
// ---------------------------------------------------------------------------
template<bool OUT_BF16, bool OUT_TRANS, bool HAS_BIAS>
__global__ __launch_bounds__(256) void gemm_bf16(
    const uint16_t* __restrict__ A, const uint16_t* __restrict__ Bt,
    void* __restrict__ Cp, const float* __restrict__ bias,
    int K, int lda, int ldb, int ldc,
    long strideA, long strideB, long strideC, float scale)
{
  __shared__ __align__(16) uint16_t lA[2][128 * LPAD];
  __shared__ __align__(16) uint16_t lB[2][128 * LPAD];

  const int tid = threadIdx.x;
  const int w   = tid >> 5;
  const int wr  = w & 3;          // wave row group (4 x 32 rows)
  const int wc  = w >> 2;         // wave col group (2 x 64 cols)
  const int bm  = blockIdx.y * 128;
  const int bn  = blockIdx.x * 128;
  const long z  = blockIdx.z;

  A  += z * strideA;
  Bt += z * strideB;

  CFrag acc[2][4];
#pragma unroll
  for (int mi = 0; mi < 2; ++mi)
#pragma unroll
    for (int ni = 0; ni < 4; ++ni)
#pragma unroll
      for (int i = 0; i < 8; ++i) acc[mi][ni].f[i] = 0.f;

  // Staging: thread s covers (row = s>>2, kgroup = s&3), two slots each.
  auto loadA = [&](int buf, int k0) {
#pragma unroll
    for (int i = 0; i < 2; ++i) {
      const int s = tid + i * 256;
      const int r = s >> 2, kg = s & 3;
      const uint4 v = *(const uint4*)(A + (long)(bm + r) * lda + k0 + kg * 8);
      *(uint4*)(&lA[buf][r * LPAD + kg * 8]) = v;
    }
  };
  auto loadB = [&](int buf, int k0) {
#pragma unroll
    for (int i = 0; i < 2; ++i) {
      const int s = tid + i * 256;
      const int n = s >> 2, kg = s & 3;
      const uint4 v = *(const uint4*)(Bt + (long)(bn + n) * ldb + k0 + kg * 8);
      *(uint4*)(&lB[buf][n * LPAD + kg * 8]) = v;
    }
  };

  const int nk = K >> 5;
  loadA(0, 0); loadB(0, 0);
  __syncthreads();

  int buf = 0;
  for (int t = 0; t < nk; ++t) {
    if (t + 1 < nk) { loadA(buf ^ 1, (t + 1) * 32); loadB(buf ^ 1, (t + 1) * 32); }
    if (t + 2 < nk) {  // gfx1250 global_prefetch_b8 for the tile after next
      __builtin_prefetch(A  + (long)(bm + (tid >> 1)) * lda + (t + 2) * 32, 0, 1);
      __builtin_prefetch(Bt + (long)(bn + (tid >> 1)) * ldb + (t + 2) * 32, 0, 1);
    }
    v16bf af0 = fragA(&lA[buf][(wr * 32 +  0) * LPAD]);
    v16bf af1 = fragA(&lA[buf][(wr * 32 + 16) * LPAD]);
#pragma unroll
    for (int ni = 0; ni < 4; ++ni) {
      v16bf bfr = fragB(&lB[buf][(wc * 64 + ni * 16) * LPAD]);
      acc[0][ni].v = __builtin_amdgcn_wmma_f32_16x16x32_bf16(
          false, af0, false, bfr, (short)0, acc[0][ni].v, false, false);
      acc[1][ni].v = __builtin_amdgcn_wmma_f32_16x16x32_bf16(
          false, af1, false, bfr, (short)0, acc[1][ni].v, false, false);
    }
    if (t + 1 < nk) { __syncthreads(); buf ^= 1; }
  }

  // C layout: lane VGPR i -> row = i + (lane<16 ? 0 : 8), col = lane&15
  const int lane = tid & 31;
  const int rsub = (lane & 16) ? 8 : 0;
  const int csub = lane & 15;
#pragma unroll
  for (int mi = 0; mi < 2; ++mi) {
#pragma unroll
    for (int ni = 0; ni < 4; ++ni) {
      const int cn  = bn + wc * 64 + ni * 16 + csub;
      const long cr0 = (long)(bm + wr * 32 + mi * 16 + rsub);
      const float bv = HAS_BIAS ? bias[cn] : 0.f;
      if (OUT_TRANS) {            // bf16, [N,M]: 8 M-values contiguous -> 16B
        Pack8 pk;
#pragma unroll
        for (int i = 0; i < 8; ++i) pk.u[i] = f2bf(acc[mi][ni].f[i] * scale + bv);
        *(uint4*)((uint16_t*)Cp + z * strideC + (long)cn * ldc + cr0) = pk.q;
      } else {
#pragma unroll
        for (int i = 0; i < 8; ++i) {
          const float val = acc[mi][ni].f[i] * scale + bv;
          if (OUT_BF16)
            ((uint16_t*)Cp)[z * strideC + (cr0 + i) * ldc + cn] = f2bf(val);
          else
            ((float*)Cp)[z * strideC + (cr0 + i) * ldc + cn] = val;
        }
      }
    }
  }
}

// ---------------------------------------------------------------------------
__global__ __launch_bounds__(256) void cvt_f32_bf16(const float* __restrict__ in,
                                                    uint16_t* __restrict__ out, long n) {
  for (long i = (long)blockIdx.x * 256 + threadIdx.x; i < n; i += (long)gridDim.x * 256)
    out[i] = f2bf(in[i]);
}

// fp32 [K,N] -> bf16 [N,K] (tiled transpose), batched over grid.z.
__global__ __launch_bounds__(256) void cvt_transpose(const float* __restrict__ in,
                                                     uint16_t* __restrict__ out,
                                                     int K, int N,
                                                     long strideIn, long strideOut) {
  __shared__ uint16_t tile[32][40];
  const long z = blockIdx.z;
  in  += z * strideIn;
  out += z * strideOut;
  const int bn = blockIdx.x * 32, bk = blockIdx.y * 32;
  const int tx = threadIdx.x & 31, ty = threadIdx.x >> 5;
#pragma unroll
  for (int i = 0; i < 32; i += 8)
    tile[ty + i][tx] = f2bf(in[(long)(bk + ty + i) * N + bn + tx]);
  __syncthreads();
#pragma unroll
  for (int i = 0; i < 32; i += 8)
    out[(long)(bn + ty + i) * K + bk + tx] = tile[tx][ty + i];
}

// Softmax over rows of length T (=1024), fp32 in -> bf16 out.
__global__ __launch_bounds__(256) void softmax_rows(const float* __restrict__ S,
                                                    uint16_t* __restrict__ P) {
  __shared__ float red[256];
  const int tid = threadIdx.x;
  const long base = (long)blockIdx.x * Tt;
  float v[4], m = -3.4e38f;
#pragma unroll
  for (int i = 0; i < 4; ++i) { v[i] = S[base + tid + i * 256]; m = fmaxf(m, v[i]); }
  red[tid] = m; __syncthreads();
#pragma unroll
  for (int off = 128; off > 0; off >>= 1) {
    if (tid < off) red[tid] = fmaxf(red[tid], red[tid + off]);
    __syncthreads();
  }
  m = red[0]; __syncthreads();
  float s = 0.f;
#pragma unroll
  for (int i = 0; i < 4; ++i) { v[i] = __expf(v[i] - m); s += v[i]; }
  const float inv = 1.f / block_reduce_sum(s, red);
#pragma unroll
  for (int i = 0; i < 4; ++i) P[base + tid + i * 256] = f2bf(v[i] * inv);
}

// y1 = LN(mha + x); emit fp32 (residual) and bf16 (FFN GEMM A operand).
__global__ __launch_bounds__(256) void add_ln(const float* __restrict__ a,
                                              const float* __restrict__ x,
                                              const float* __restrict__ g,
                                              const float* __restrict__ be,
                                              float* __restrict__ y,
                                              uint16_t* __restrict__ yb) {
  __shared__ float red[256];
  const int tid = threadIdx.x;
  const long base = (long)blockIdx.x * Dd;
  float v0 = a[base + tid] + x[base + tid];
  float v1 = a[base + tid + 256] + x[base + tid + 256];
  const float mu = block_reduce_sum(v0 + v1, red) * (1.f / Dd);
  const float t0 = v0 - mu, t1 = v1 - mu;
  const float var = block_reduce_sum(t0 * t0 + t1 * t1, red) * (1.f / Dd);
  const float rs = rsqrtf(var + 1e-5f);
  const float o0 = t0 * rs * g[tid] + be[tid];
  const float o1 = t1 * rs * g[tid + 256] + be[tid + 256];
  y[base + tid] = o0;         y[base + tid + 256] = o1;
  yb[base + tid] = f2bf(o0);  yb[base + tid + 256] = f2bf(o1);
}

// out = LN(relu(ff) + y1)
__global__ __launch_bounds__(256) void relu_add_ln(const float* __restrict__ ff,
                                                   const float* __restrict__ y1,
                                                   const float* __restrict__ g,
                                                   const float* __restrict__ be,
                                                   float* __restrict__ out) {
  __shared__ float red[256];
  const int tid = threadIdx.x;
  const long base = (long)blockIdx.x * Dd;
  float v0 = fmaxf(ff[base + tid], 0.f) + y1[base + tid];
  float v1 = fmaxf(ff[base + tid + 256], 0.f) + y1[base + tid + 256];
  const float mu = block_reduce_sum(v0 + v1, red) * (1.f / Dd);
  const float t0 = v0 - mu, t1 = v1 - mu;
  const float var = block_reduce_sum(t0 * t0 + t1 * t1, red) * (1.f / Dd);
  const float rs = rsqrtf(var + 1e-5f);
  out[base + tid]       = t0 * rs * g[tid] + be[tid];
  out[base + tid + 256] = t1 * rs * g[tid + 256] + be[tid + 256];
}

// ---------------------------------------------------------------------------
extern "C" void kernel_launch(void* const* d_in, const int* in_sizes, int n_in,
                              void* d_out, int out_size, void* d_ws, size_t ws_size,
                              hipStream_t stream) {
  (void)in_sizes; (void)n_in; (void)out_size; (void)ws_size;
  const float* x    = (const float*)d_in[0];
  const float* Wq   = (const float*)d_in[1];
  const float* bq   = (const float*)d_in[2];
  const float* Wk   = (const float*)d_in[3];
  const float* bk   = (const float*)d_in[4];
  const float* Wv   = (const float*)d_in[5];
  const float* bv   = (const float*)d_in[6];
  const float* Wo   = (const float*)d_in[7];
  const float* bo   = (const float*)d_in[8];
  const float* ln1g = (const float*)d_in[9];
  const float* ln1b = (const float*)d_in[10];
  const float* W1   = (const float*)d_in[11];
  const float* b1   = (const float*)d_in[12];
  const float* W2   = (const float*)d_in[13];
  const float* b2   = (const float*)d_in[14];
  const float* ln2g = (const float*)d_in[15];
  const float* ln2b = (const float*)d_in[16];

  char* ws = (char*)d_ws;
  const long MB = 1l << 20;
  // persistent bf16 copies (all weights transposed to [N,K])
  uint16_t* xb  = (uint16_t*)(ws + 0 * MB);    //  8 MB  [8192,512]
  uint16_t* Wqt = (uint16_t*)(ws + 8 * MB);    //  4 MB  [H][512,512]
  uint16_t* Wkt = (uint16_t*)(ws + 12 * MB);   //  4 MB
  uint16_t* Wvt = (uint16_t*)(ws + 16 * MB);   //  4 MB
  uint16_t* Wot = (uint16_t*)(ws + 20 * MB);   //  4 MB  [512,4096]
  uint16_t* W1t = (uint16_t*)(ws + 24 * MB);   //  2 MB  [2048,512]
  uint16_t* W2t = (uint16_t*)(ws + 26 * MB);   //  2 MB  [512,2048]
  // per-head buffers (reused across heads, aliased after the head loop)
  uint16_t* Qh  = (uint16_t*)(ws + 28 * MB);   //  8 MB  [B*T, D] bf16
  uint16_t* Kh  = (uint16_t*)(ws + 36 * MB);   //  8 MB  [B*T, D]  ([N,K] for S!)
  uint16_t* Vt  = (uint16_t*)(ws + 44 * MB);   //  8 MB  [D, B*T]  transposed
  float*    S   = (float*)   (ws + 52 * MB);   // 32 MB  [B,T,T] fp32
  uint16_t* P   = (uint16_t*)(ws + 84 * MB);   // 16 MB  [B,T,T] bf16
  uint16_t* CC  = (uint16_t*)(ws + 100 * MB);  // 64 MB  concat [8192,4096] bf16
  // aliases used after the head loop (donors are dead by then)
  float*    mha = (float*)   (ws + 28 * MB);   // 16 MB  (over Qh/Kh)
  float*    y1  = (float*)   (ws + 52 * MB);   // 16 MB  (over S)
  uint16_t* y1b = (uint16_t*)(ws + 68 * MB);   //  8 MB  (over S)
  uint16_t* h1  = (uint16_t*)(ws + 100 * MB);  // 32 MB  (over CC, dead post-Wo)
  float*    ff  = (float*)   (ws + 132 * MB);  // 16 MB

  const dim3 blk(256);
  auto grid = [](int M, int N, int z) { return dim3(N / 128, M / 128, z); };

  // --- bf16 conversions (weights transposed to [N,K]) ---
  cvt_f32_bf16<<<2048, blk, 0, stream>>>(x, xb, (long)Mrows * Dd);
  cvt_transpose<<<dim3(Dd / 32, Dd / 32, Hh), blk, 0, stream>>>(
      Wq, Wqt, Dd, Dd, (long)Dd * Dd, (long)Dd * Dd);
  cvt_transpose<<<dim3(Dd / 32, Dd / 32, Hh), blk, 0, stream>>>(
      Wk, Wkt, Dd, Dd, (long)Dd * Dd, (long)Dd * Dd);
  cvt_transpose<<<dim3(Dd / 32, Dd / 32, Hh), blk, 0, stream>>>(
      Wv, Wvt, Dd, Dd, (long)Dd * Dd, (long)Dd * Dd);
  cvt_transpose<<<dim3(Dd / 32, (Hh * Dd) / 32, 1), blk, 0, stream>>>(
      Wo, Wot, Hh * Dd, Dd, 0, 0);
  cvt_transpose<<<dim3(DFFd / 32, Dd / 32, 1), blk, 0, stream>>>(
      W1, W1t, Dd, DFFd, 0, 0);
  cvt_transpose<<<dim3(Dd / 32, DFFd / 32, 1), blk, 0, stream>>>(
      W2, W2t, DFFd, Dd, 0, 0);

  const float sscale = 0.04419417382415922f;  // 512^-0.5

  // --- attention, one head at a time (stream-ordered buffer reuse) ---
  for (int h = 0; h < Hh; ++h) {
    const long wOff = (long)h * Dd * Dd;
    gemm_bf16<true, false, true><<<grid(Mrows, Dd, 1), blk, 0, stream>>>(
        xb, Wqt + wOff, Qh, bq + h * Dd, Dd, Dd, Dd, Dd, 0, 0, 0, 1.f);
    gemm_bf16<true, false, true><<<grid(Mrows, Dd, 1), blk, 0, stream>>>(
        xb, Wkt + wOff, Kh, bk + h * Dd, Dd, Dd, Dd, Dd, 0, 0, 0, 1.f);
    // V projection stored transposed: Vt[d, b*T+t]  (ldc = Mrows)
    gemm_bf16<true, true, true><<<grid(Mrows, Dd, 1), blk, 0, stream>>>(
        xb, Wvt + wOff, Vt, bv + h * Dd, Dd, Dd, Dd, Mrows, 0, 0, 0, 1.f);
    // S[b] = scale * Q[b] @ K[b]^T   (K is [T,D] = [N,K]-major already)
    gemm_bf16<false, false, false><<<grid(Tt, Tt, Bb), blk, 0, stream>>>(
        Qh, Kh, S, nullptr, Dd, Dd, Dd, Tt,
        (long)Tt * Dd, (long)Tt * Dd, (long)Tt * Tt, sscale);
    softmax_rows<<<Bb * Tt, blk, 0, stream>>>(S, P);
    // concat[b*T.., h*D..] = P[b] @ V[b]; B operand = Vt rows d, k = s
    gemm_bf16<true, false, false><<<grid(Tt, Dd, Bb), blk, 0, stream>>>(
        P, Vt, CC + h * Dd, nullptr, Tt, Tt, Mrows, Hh * Dd,
        (long)Tt * Tt, (long)Tt, (long)Tt * Hh * Dd, 1.f);
  }

  // --- output projection + residual + LN1 ---
  gemm_bf16<false, false, true><<<grid(Mrows, Dd, 1), blk, 0, stream>>>(
      CC, Wot, mha, bo, Hh * Dd, Hh * Dd, Hh * Dd, Dd, 0, 0, 0, 1.f);
  add_ln<<<Mrows, blk, 0, stream>>>(mha, x, ln1g, ln1b, y1, y1b);

  // --- FFN: relu((y1 W1 + b1) W2 + b2) + y1, then LN2 ---
  gemm_bf16<true, false, true><<<grid(Mrows, DFFd, 1), blk, 0, stream>>>(
      y1b, W1t, h1, b1, Dd, Dd, Dd, DFFd, 0, 0, 0, 1.f);
  gemm_bf16<false, false, true><<<grid(Mrows, Dd, 1), blk, 0, stream>>>(
      h1, W2t, ff, b2, DFFd, DFFd, DFFd, Dd, 0, 0, 0, 1.f);
  relu_add_ln<<<Mrows, blk, 0, stream>>>(ff, y1, ln2g, ln2b, (float*)d_out);
}
